// RelativePositionMultiAttention_86328842649868
// MI455X (gfx1250) — matrix operand order repaired and checked by reading, not compiled, not run on hardware
//
#include <hip/hip_runtime.h>
#include <math.h>

#define B_ 1024
#define S_ 100
#define D_ 512
#define HALF_ 256
#define SP_ 112            // S padded to 7 tiles of 16
#define KP_ 128            // S padded to 4 K-steps of 32 for PV
#define NEG_BIG_F (-4294967295.0f)
#define PAD_NEG   (-3.0e38f)

typedef __attribute__((ext_vector_type(16))) __bf16        v16bf;
typedef __attribute__((ext_vector_type(8)))  __bf16        v8bf;
typedef __attribute__((ext_vector_type(8)))  float         v8f;
typedef __attribute__((ext_vector_type(8)))  unsigned int  v8u;
typedef __attribute__((ext_vector_type(4)))  unsigned int  v4u;
typedef __attribute__((ext_vector_type(4)))  float         v4f;

// ---------- helpers ----------
// native conversion -> v_cvt_pk_bf16_f32 (RTNE)
__device__ __forceinline__ __bf16 f2bf(float f) { return (__bf16)f; }
__device__ __forceinline__ unsigned short f2bfu(float f) {
  return __builtin_bit_cast(unsigned short, (__bf16)f);
}
__device__ __forceinline__ float bf2f(unsigned short s) {
  return __builtin_bit_cast(float, ((unsigned int)s) << 16);
}
__device__ __forceinline__ v16bf as_v16bf(v8u x) { return __builtin_bit_cast(v16bf, x); }
__device__ __forceinline__ v16bf zero_v16bf() {
  v8u z;
#pragma unroll
  for (int i = 0; i < 8; ++i) z[i] = 0u;
  return as_v16bf(z);
}
__device__ __forceinline__ v8f zero_v8f() {
  v8f z;
#pragma unroll
  for (int i = 0; i < 8; ++i) z[i] = 0.f;
  return z;
}
// concat two 16B chunks (K kb..kb+7 and K kb+8..kb+15) into an A/B operand
__device__ __forceinline__ v16bf bf_cat(v4u lo, v4u hi) {
  v8u t;
#pragma unroll
  for (int i = 0; i < 4; ++i) { t[i] = lo[i]; t[i + 4] = hi[i]; }
  return as_v16bf(t);
}
// A operand built from f32 global memory (rows of X), ks {0..15} rel. to p
__device__ __forceinline__ v16bf a_from_f32(const float* p) {
  v4f f0 = *(const v4f*)(p + 0);
  v4f f1 = *(const v4f*)(p + 4);
  v4f f2 = *(const v4f*)(p + 8);
  v4f f3 = *(const v4f*)(p + 12);
  v16bf A;
#pragma unroll
  for (int i = 0; i < 4; ++i) {
    A[i]      = f2bf(f0[i]);
    A[4 + i]  = f2bf(f1[i]);
    A[8 + i]  = f2bf(f2[i]);
    A[12 + i] = f2bf(f3[i]);
  }
  return A;
}
__device__ __forceinline__ v4u pack_bf8(v8f a) {
  v8bf h;
#pragma unroll
  for (int i = 0; i < 8; ++i) h[i] = f2bf(a[i]);
  return __builtin_bit_cast(v4u, h);
}
__device__ __forceinline__ v8f wmma_bf16(v16bf a, v16bf b, v8f c) {
  return __builtin_amdgcn_wmma_f32_16x16x32_bf16(false, a, false, b, (short)0, c, false, false);
}

// ---------- LDS layout (dynamic, ~254 KB of the 320 KB WGP LDS) ----------
#define LDS_KEFF 0                 // 112 x 256 bf16 = 57344
#define LDS_Q    57344             // 112 x 256 bf16 = 57344
#define LDS_VT   114688            // 256 x 128 bf16 = 65536 (V transposed)
#define LDS_PBF  180224            // 112 x 128 bf16 = 28672 (softmax probs)
#define LDS_SC   208896            // 112 x 112 f32  = 50176 (scores)
#define LDS_BIAS 259072            // 112 f32
#define LDS_TOTAL 259584

// ---------- prep: transpose+convert weights to bf16 ----------
__global__ void prep_weights(const float* __restrict__ Wq, const float* __restrict__ Wv,
                             unsigned short* __restrict__ WqT, unsigned short* __restrict__ WvT) {
  int idx = blockIdx.x * 256 + threadIdx.x;        // 2 * 256*512 elements
  const float* W = (idx < HALF_ * D_) ? Wq : Wv;
  unsigned short* WT = (idx < HALF_ * D_) ? WqT : WvT;
  int r = idx & (HALF_ * D_ - 1);
  int h = r >> 9, k = r & 511;                     // WT[h][k] = W[k][h]
  WT[h * D_ + k] = f2bfu(W[k * HALF_ + h]);
}

// ---------- prep: positional sums + positional bias ----------
__global__ void prep_pos(const float* __restrict__ v,
                         unsigned short* __restrict__ Psum, float* __restrict__ biasP) {
  int t = blockIdx.x, h = threadIdx.x;             // 100 blocks x 256 threads
  float scale = __expf((float)h * (-(logf(10000.f) / (float)(HALF_ - 1))));
  float ang = (float)(t - S_ / 2) * scale;
  float sn = __sinf(ang), cs = __cosf(ang);
  Psum[t * HALF_ + h] = f2bfu(sn + cs);
  __shared__ float red[256];
  red[h] = sn * v[h] + cs * v[HALF_ + h];
  __syncthreads();
  for (int o = 128; o > 0; o >>= 1) { if (h < o) red[h] += red[h + o]; __syncthreads(); }
  if (h == 0) biasP[t] = red[0];
}

// ---------- stream: out[..., :512] = inputs ----------
__global__ void copy_inputs(const float* __restrict__ x, float* __restrict__ out) {
  int idx = blockIdx.x * 256 + threadIdx.x;        // B*S*128 float4's
  int c = idx & 127;
  long bs = idx >> 7;
  v4f val = *(const v4f*)(x + bs * (long)D_ + c * 4);
  *(v4f*)(out + bs * (long)(D_ + HALF_) + c * 4) = val;
}

// ---------- main: one workgroup (8 waves) per batch ----------
__global__ void attn_kernel(const float* __restrict__ X, const int* __restrict__ masks,
                            const float* __restrict__ u,
                            const unsigned short* __restrict__ WqT,
                            const unsigned short* __restrict__ WvT,
                            const unsigned short* __restrict__ Psum,
                            const float* __restrict__ biasP,
                            float* __restrict__ out) {
  extern __shared__ char smem[];
  unsigned short* KEFF = (unsigned short*)(smem + LDS_KEFF);
  unsigned short* QL   = (unsigned short*)(smem + LDS_Q);
  unsigned short* VT   = (unsigned short*)(smem + LDS_VT);
  unsigned short* PBF  = (unsigned short*)(smem + LDS_PBF);
  float*          SC   = (float*)(smem + LDS_SC);
  float*          BIAS = (float*)(smem + LDS_BIAS);

  const int b = blockIdx.x;
  const int tid = threadIdx.x;
  const int wave = tid >> 5, lane = tid & 31;
  const int mrow = lane & 15;                 // row (A) or col (B, C/D) within tile
  const int hi8  = (lane >> 4) * 8;           // C/D row offset for lanes 16-31
  const int kb   = (lane >> 4) * 16;          // A-operand K base
  const int kk16 = (lane >> 4) * 16;          // B-operand K base
  const float* Xb = X + (long)b * S_ * D_;

  // ---- phase 1: Keff = K1+K2+Psum (bf16), bias[t] = K·u + biasP + mask ----
  for (int idx = tid; idx < SP_ * HALF_; idx += 256) {
    int t = idx >> 8, h = idx & 255;
    float val = 0.f;
    if (t < S_)
      val = Xb[t * D_ + h] + Xb[t * D_ + HALF_ + h] + bf2f(Psum[t * HALF_ + h]);
    KEFF[idx] = f2bfu(val);
  }
  for (int t = wave; t < SP_; t += 8) {
    if (t < S_) {
      const float* xr = Xb + t * D_;
      float acc = 0.f;
      for (int h = lane; h < HALF_; h += 32)
        acc += xr[h] * u[h] + xr[HALF_ + h] * u[HALF_ + h];
      for (int off = 16; off > 0; off >>= 1) acc += __shfl_down(acc, off, 32);
      if (lane == 0)
        BIAS[t] = acc + biasP[t] + (float)masks[b * S_ + t] * NEG_BIG_F;
    } else if (lane == 0) {
      BIAS[t] = PAD_NEG;
    }
  }

  // ---- phase 2: Q = Xb*Wq, V = Xb*Wv via WMMA bf16; V stored transposed ----
  // M loop peeled: tiles 0..5 have all rows < 100 (no guard, no exec churn);
  // tile 6 (rows 96..111) takes the guarded path.
  {
    const int n0 = wave * 32;                       // two 16-wide N columns per wave
    const unsigned short* wq0 = WqT + (long)(n0 + mrow) * D_ + kk16;
    const unsigned short* wq1 = WqT + (long)(n0 + 16 + mrow) * D_ + kk16;
    const unsigned short* wv0 = WvT + (long)(n0 + mrow) * D_ + kk16;
    const unsigned short* wv1 = WvT + (long)(n0 + 16 + mrow) * D_ + kk16;
    __builtin_prefetch(wq0, 0, 0);                  // global_prefetch_b8
    __builtin_prefetch(wv0, 0, 0);

    auto qv_tile = [&](int m, bool guard) {
      int s0 = m * 16;
      int s = s0 + mrow;
      const float* xp = Xb + (long)s * D_ + kb;
      v8f aq0 = zero_v8f(), aq1 = zero_v8f(), av0 = zero_v8f(), av1 = zero_v8f();
#pragma unroll 4
      for (int kt = 0; kt < 16; ++kt) {
        int k0 = kt * 32;
        v16bf A = (!guard || s < S_) ? a_from_f32(xp + k0) : zero_v16bf();
        v16bf Bq0 = as_v16bf(*(const v8u*)(wq0 + k0));
        v16bf Bq1 = as_v16bf(*(const v8u*)(wq1 + k0));
        v16bf Bv0 = as_v16bf(*(const v8u*)(wv0 + k0));
        v16bf Bv1 = as_v16bf(*(const v8u*)(wv1 + k0));
        aq0 = wmma_bf16(A, Bq0, aq0);
        aq1 = wmma_bf16(A, Bq1, aq1);
        av0 = wmma_bf16(A, Bv0, av0);
        av1 = wmma_bf16(A, Bv1, av1);
      }
      // Q row-major
#pragma unroll
      for (int r = 0; r < 8; ++r) {
        int sr = s0 + r + hi8;
        QL[sr * HALF_ + n0 + mrow]      = f2bfu(aq0[r]);
        QL[sr * HALF_ + n0 + 16 + mrow] = f2bfu(aq1[r]);
      }
      // V transposed: VT[h][s], 8 consecutive s per lane -> one 16B store
      *(v4u*)(VT + (n0 + mrow) * KP_ + s0 + hi8)      = pack_bf8(av0);
      *(v4u*)(VT + (n0 + 16 + mrow) * KP_ + s0 + hi8) = pack_bf8(av1);
    };
    for (int m = 0; m < 6; ++m) qv_tile(m, false);
    qv_tile(6, true);
  }
  __syncthreads();

  // ---- phase 3: scores = Q * Keff^T (7x7 tiles, K=256) ----
  for (int tile = wave; tile < 49; tile += 8) {
    int s0 = (tile / 7) * 16, t0 = (tile % 7) * 16;
    v8f acc = zero_v8f();
#pragma unroll
    for (int kt = 0; kt < 8; ++kt) {
      int k0 = kt * 32;
      const unsigned short* qp = QL + (s0 + mrow) * HALF_ + k0 + kb;
      v16bf A = bf_cat(*(const v4u*)qp, *(const v4u*)(qp + 8));
      v16bf Bt = as_v16bf(*(const v8u*)(KEFF + (t0 + mrow) * HALF_ + k0 + kk16));
      acc = wmma_bf16(A, Bt, acc);
    }
#pragma unroll
    for (int r = 0; r < 8; ++r)
      SC[(s0 + r + hi8) * SP_ + t0 + mrow] = acc[r];
  }
  __syncthreads();

  // ---- phase 4: softmax over t (bias folded in), write P as bf16 [112][128] ----
  if (tid < SP_) {
    float* sr = SC + tid * SP_;
    float mx = -3.4e38f;
    for (int t = 0; t < SP_; ++t) {
      float s = sr[t] + BIAS[t];
      sr[t] = s;
      mx = fmaxf(mx, s);
    }
    float sum = 0.f;
    for (int t = 0; t < SP_; ++t) {
      float e = __expf(sr[t] - mx);
      sr[t] = e;
      sum += e;
    }
    float inv = 1.f / sum;
    unsigned short* pr = PBF + tid * KP_;
    for (int t = 0; t < SP_; ++t) pr[t] = f2bfu(sr[t] * inv);
    for (int t = SP_; t < KP_; ++t) pr[t] = 0;
  }
  __syncthreads();

  // ---- phase 5: result = P * V (via VT), store to out[..., 512:768] ----
  {
    const int n0 = wave * 32;
    auto pv_tile = [&](int m, bool guard) {
      int s0 = m * 16;
      v8f a0 = zero_v8f(), a1 = zero_v8f();
#pragma unroll
      for (int kt = 0; kt < 4; ++kt) {
        int k0 = kt * 32;
        const unsigned short* pp = PBF + (s0 + mrow) * KP_ + k0 + kb;
        v16bf A = bf_cat(*(const v4u*)pp, *(const v4u*)(pp + 8));
        v16bf B0 = as_v16bf(*(const v8u*)(VT + (n0 + mrow) * KP_ + k0 + kk16));
        v16bf B1 = as_v16bf(*(const v8u*)(VT + (n0 + 16 + mrow) * KP_ + k0 + kk16));
        a0 = wmma_bf16(A, B0, a0);
        a1 = wmma_bf16(A, B1, a1);
      }
#pragma unroll
      for (int r = 0; r < 8; ++r) {
        int s = s0 + r + hi8;
        if (!guard || s < S_) {
          float* orow = out + ((long)b * S_ + s) * (D_ + HALF_) + D_;
          orow[n0 + mrow]      = a0[r];
          orow[n0 + 16 + mrow] = a1[r];
        }
      }
    };
    for (int m = 0; m < 6; ++m) pv_tile(m, false);
    pv_tile(6, true);
  }
}

extern "C" void kernel_launch(void* const* d_in, const int* in_sizes, int n_in,
                              void* d_out, int out_size, void* d_ws, size_t ws_size,
                              hipStream_t stream) {
  (void)in_sizes; (void)n_in; (void)out_size; (void)ws_size;
  const float* x     = (const float*)d_in[0];
  const int*   masks = (const int*)d_in[1];
  const float* Wq    = (const float*)d_in[2];
  const float* Wv    = (const float*)d_in[3];
  const float* u     = (const float*)d_in[4];
  const float* v     = (const float*)d_in[5];
  float* out = (float*)d_out;

  char* ws = (char*)d_ws;
  unsigned short* WqT  = (unsigned short*)ws;                       // 256*512 bf16
  unsigned short* WvT  = WqT + HALF_ * D_;                          // 256*512 bf16
  unsigned short* Psum = WvT + HALF_ * D_;                          // 100*256 bf16
  float* biasP = (float*)(ws + (size_t)(2 * HALF_ * D_ + S_ * HALF_) * 2);

  prep_weights<<<(2 * HALF_ * D_) / 256, 256, 0, stream>>>(Wq, Wv, WqT, WvT);
  prep_pos<<<S_, 256, 0, stream>>>(v, Psum, biasP);
  copy_inputs<<<(B_ * S_ * (D_ / 4)) / 256, 256, 0, stream>>>(x, out);
  attn_kernel<<<B_, 256, LDS_TOTAL, stream>>>(x, masks, u, WqT, WvT, Psum, biasP, out);
}
